// CSGNN_26611617366361
// MI455X (gfx1250) — matrix operand order repaired and checked by reading, not compiled, not run on hardware
//
#include <hip/hip_runtime.h>
#include <hip/hip_bf16.h>

// ---------------------------------------------------------------------------
// CSGNN forward for MI455X (gfx1250, wave32).
// GEMMs run as bf16 WMMA (v_wmma_f32_16x16x32_bf16) with f32 accumulate.
// Intermediate activations (T, x1, FEAT, RT) are stored as bf16 -- identical
// precision to what WMMA would consume anyway, at half the memory traffic.
// ---------------------------------------------------------------------------

#define NN   50000      // nodes
#define EE   400000     // edges
#define FD   256        // feature dim (also 2*H1 = 2*H2)
#define PP   100000     // pairs
#define D1D  512        // decoder hidden

typedef __bf16 v16bf __attribute__((ext_vector_type(16)));
typedef __bf16 v8bf  __attribute__((ext_vector_type(8)));
typedef float  v8f   __attribute__((ext_vector_type(8)));

__device__ __forceinline__ void atomic_add_f32(float* p, float v) {
  __hip_atomic_fetch_add(p, v, __ATOMIC_RELAXED, __HIP_MEMORY_SCOPE_AGENT);
}

// ---------------------------------------------------------------------------
// Weight convert + transpose: W[K, Nout] f32 (row major)  ->  Wt[Nout, K] bf16
// ---------------------------------------------------------------------------
__global__ __launch_bounds__(256) void transpose_bf16_kernel(
    const float* __restrict__ W, __bf16* __restrict__ Wt, int K, int Nout) {
  int idx = blockIdx.x * 256 + threadIdx.x;
  if (idx >= K * Nout) return;
  int k = idx / Nout;
  int n = idx - k * Nout;
  Wt[(size_t)n * K + k] = (__bf16)W[idx];
}

// ---------------------------------------------------------------------------
// H = (1 + eps) * X          (elementwise, eps is a device scalar)
// ---------------------------------------------------------------------------
template <typename TX>
__global__ __launch_bounds__(256) void scale_copy_kernel(
    float* __restrict__ H, const TX* __restrict__ X,
    const float* __restrict__ eps, int n) {
  int i = blockIdx.x * 256 + threadIdx.x;
  if (i >= n) return;
  H[i] = (1.0f + eps[0]) * (float)X[i];
}

// ---------------------------------------------------------------------------
// Edge scatter: H[dst[e], :] += X[src[e], :]    (one edge per 256-thread block)
// ---------------------------------------------------------------------------
template <typename TX>
__global__ __launch_bounds__(256) void edge_scatter_kernel(
    float* __restrict__ H, const TX* __restrict__ X,
    const int* __restrict__ src, const int* __restrict__ dst, int E) {
  int e = blockIdx.x;
  int c = threadIdx.x;                         // 256 feature columns
  int s = src[e], d = dst[e];
  float v = (float)X[(size_t)s * FD + c];
  atomic_add_f32(&H[(size_t)d * FD + c], v);
}

// ---------------------------------------------------------------------------
// WMMA GEMM: C[M, Nout] = act(A[M, K] @ Wt^T + bias), Wt is bf16 [Nout, K].
// TA = float (convert on load) or __bf16 (direct b128 loads, no cvt).
// TC = float or __bf16 output.
// Each wave computes a 16x64 output tile (one A frag reused across 4 B frags).
// Block = 256 threads = 8 waves = 128 rows x 64 cols.
// Grid = (ceil(M/128), Nout/64).  K % 32 == 0, Nout % 64 == 0.
// ---------------------------------------------------------------------------
template <bool RELU, typename TA, typename TC>
__global__ __launch_bounds__(256) void wmma_gemm_kernel(
    const TA*  __restrict__ A, const __bf16* __restrict__ Wt,
    const float* __restrict__ bias, TC* __restrict__ C,
    int M, int K, int Nout, int ldc) {
  const int wave = threadIdx.x >> 5;
  const int lane = threadIdx.x & 31;
  const int half = lane >> 4;      // 0: lanes 0-15, 1: lanes 16-31
  const int l15  = lane & 15;

  const int rowBase = blockIdx.x * 128 + wave * 16;
  const int colBase = blockIdx.y * 64;
  if (rowBase >= M) return;        // uniform per wave -> EXEC stays all-ones

  // A fragment source row (clamped; out-of-range rows masked at store)
  int arow = rowBase + l15;
  if (arow >= M) arow = M - 1;
  const TA* Ap = A + (size_t)arow * K;

  // B fragment base: lane l15 holds column n = colBase + l15 (+16j per subtile),
  // element e: k = 16*(e>>3) + 8*half + (e&7) within the 32-wide K slab.
  const __bf16* Bp = Wt + (size_t)(colBase + l15) * K + half * 8;

  v8f acc0 = {}, acc1 = {}, acc2 = {}, acc3 = {};

  for (int kb = 0; kb < K; kb += 32) {
    // ---- A fragment: ISA 16-bit A layout (16x32, MxK) ----
    // element e: k = kb + 16*(e>>3) + 8*half + (e&7)
    v16bf a;
    if constexpr (__is_same(TA, float)) {
      const float4 a0 = *(const float4*)(Ap + kb + 8 * half);
      const float4 a1 = *(const float4*)(Ap + kb + 8 * half + 4);
      const float4 a2 = *(const float4*)(Ap + kb + 16 + 8 * half);
      const float4 a3 = *(const float4*)(Ap + kb + 16 + 8 * half + 4);
      a[0]  = (__bf16)a0.x; a[1]  = (__bf16)a0.y; a[2]  = (__bf16)a0.z; a[3]  = (__bf16)a0.w;
      a[4]  = (__bf16)a1.x; a[5]  = (__bf16)a1.y; a[6]  = (__bf16)a1.z; a[7]  = (__bf16)a1.w;
      a[8]  = (__bf16)a2.x; a[9]  = (__bf16)a2.y; a[10] = (__bf16)a2.z; a[11] = (__bf16)a2.w;
      a[12] = (__bf16)a3.x; a[13] = (__bf16)a3.y; a[14] = (__bf16)a3.z; a[15] = (__bf16)a3.w;
    } else {
      const v8bf g0 = *(const v8bf*)(Ap + kb + 8 * half);
      const v8bf g1 = *(const v8bf*)(Ap + kb + 16 + 8 * half);
      a = __builtin_shufflevector(g0, g1, 0, 1, 2, 3, 4, 5, 6, 7,
                                          8, 9, 10, 11, 12, 13, 14, 15);
    }
    if (kb + 32 < K)  // prefetch next K slab of this A row
      __builtin_prefetch((const void*)(Ap + kb + 32), 0, 1);

    // ---- B fragments: two 16B bf16 loads 16 elements apart, 4 column subtiles ----
    auto ldb = [&](const __bf16* q) -> v16bf {
      const v8bf h0 = *(const v8bf*)(q);
      const v8bf h1 = *(const v8bf*)(q + 16);
      return __builtin_shufflevector(h0, h1, 0, 1, 2, 3, 4, 5, 6, 7,
                                     8, 9, 10, 11, 12, 13, 14, 15);
    };
    v16bf b0 = ldb(Bp + kb);
    v16bf b1 = ldb(Bp + (size_t)16 * K + kb);
    v16bf b2 = ldb(Bp + (size_t)32 * K + kb);
    v16bf b3 = ldb(Bp + (size_t)48 * K + kb);

    acc0 = __builtin_amdgcn_wmma_f32_16x16x32_bf16(false, a, false, b0, (short)0, acc0, false, false);
    acc1 = __builtin_amdgcn_wmma_f32_16x16x32_bf16(false, a, false, b1, (short)0, acc1, false, false);
    acc2 = __builtin_amdgcn_wmma_f32_16x16x32_bf16(false, a, false, b2, (short)0, acc2, false, false);
    acc3 = __builtin_amdgcn_wmma_f32_16x16x32_bf16(false, a, false, b3, (short)0, acc3, false, false);
  }

  // ---- store: C/D layout: lane -> n = l15 (+16j), VGPR v -> m = 8*half + v ----
  const float bv0 = bias[colBase + l15];
  const float bv1 = bias[colBase + 16 + l15];
  const float bv2 = bias[colBase + 32 + l15];
  const float bv3 = bias[colBase + 48 + l15];
#pragma unroll
  for (int v = 0; v < 8; ++v) {
    int r = rowBase + half * 8 + v;
    if (r < M) {
      TC* crow = C + (size_t)r * ldc + colBase + l15;
      float v0 = acc0[v] + bv0, v1 = acc1[v] + bv1;
      float v2 = acc2[v] + bv2, v3 = acc3[v] + bv3;
      if (RELU) {
        v0 = fmaxf(v0, 0.f); v1 = fmaxf(v1, 0.f);
        v2 = fmaxf(v2, 0.f); v3 = fmaxf(v3, 0.f);
      }
      crow[0] = (TC)v0; crow[16] = (TC)v1; crow[32] = (TC)v2; crow[48] = (TC)v3;
    }
  }
}

template <bool RELU, typename TA, typename TC>
static void launch_gemm(const TA* A, const __bf16* Wt, const float* bias, TC* C,
                        int M, int K, int Nout, int ldc, hipStream_t stream) {
  dim3 g((M + 127) / 128, Nout / 64);
  wmma_gemm_kernel<RELU, TA, TC><<<g, 256, 0, stream>>>(A, Wt, bias, C, M, K, Nout, ldc);
}

// ---------------------------------------------------------------------------
// Small helpers: zero, column sums, sigmoid + disc_w @ c
// ---------------------------------------------------------------------------
__global__ __launch_bounds__(256) void zero_kernel(float* p, int n) {
  int i = blockIdx.x * 256 + threadIdx.x;
  if (i < n) p[i] = 0.f;
}

__global__ __launch_bounds__(256) void colsum_kernel(
    const float* __restrict__ X, float* __restrict__ colsum, int rows) {
  int c = threadIdx.x;                  // 256 columns
  float s = 0.f;
  for (int r = blockIdx.x; r < rows; r += gridDim.x)
    s += X[(size_t)r * FD + c];
  atomic_add_f32(&colsum[c], s);
}

__global__ __launch_bounds__(256) void disc_v_kernel(
    const float* __restrict__ cs_os, const float* __restrict__ cs_osa,
    const float* __restrict__ disc_w,
    float* __restrict__ v_os, float* __restrict__ v_osa, float invN) {
  __shared__ float h0[FD], h1[FD];
  int t = threadIdx.x;
  h0[t] = 1.f / (1.f + __expf(-cs_os[t] * invN));
  h1[t] = 1.f / (1.f + __expf(-cs_osa[t] * invN));
  __syncthreads();
  float a = 0.f, b = 0.f;
  const float* wrow = disc_w + (size_t)t * FD;
  for (int j = 0; j < FD; ++j) { float w = wrow[j]; a += w * h0[j]; b += w * h1[j]; }
  v_os[t] = a; v_osa[t] = b;
}

// ---------------------------------------------------------------------------
// Bilinear discriminator scores: one wave per row, 4 dot products per row.
// ret_os[n]  = [x2 . v_os,  x2a . v_os ] + b
// ret_osa[n] = [x2a . v_osa, x2 . v_osa] + b
// ---------------------------------------------------------------------------
__global__ __launch_bounds__(256) void disc_scores_kernel(
    const float* __restrict__ x2, const float* __restrict__ x2a,
    const float* __restrict__ v_os, const float* __restrict__ v_osa,
    const float* __restrict__ disc_b,
    float* __restrict__ ret_os, float* __restrict__ ret_osa, int rows) {
  __shared__ float vs[FD], va[FD];
  int t = threadIdx.x;
  vs[t] = v_os[t]; va[t] = v_osa[t];
  __syncthreads();
  int wave = t >> 5, lane = t & 31;
  int row = blockIdx.x * 8 + wave;
  if (row >= rows) return;
  const float* r2  = x2  + (size_t)row * FD;
  const float* r2a = x2a + (size_t)row * FD;
  float d0 = 0.f, d1 = 0.f, d2 = 0.f, d3 = 0.f;
#pragma unroll
  for (int i = 0; i < 8; ++i) {
    int c = lane * 8 + i;
    float a = r2[c], b = r2a[c];
    d0 += a * vs[c]; d1 += a * va[c];
    d2 += b * vs[c]; d3 += b * va[c];
  }
#pragma unroll
  for (int off = 16; off > 0; off >>= 1) {
    d0 += __shfl_down(d0, off, 32);
    d1 += __shfl_down(d1, off, 32);
    d2 += __shfl_down(d2, off, 32);
    d3 += __shfl_down(d3, off, 32);
  }
  if (lane == 0) {
    float bb = disc_b[0];
    ret_os [row * 2 + 0] = d0 + bb;   // x2  . v_os
    ret_os [row * 2 + 1] = d2 + bb;   // x2a . v_os
    ret_osa[row * 2 + 0] = d3 + bb;   // x2a . v_osa
    ret_osa[row * 2 + 1] = d1 + bb;   // x2  . v_osa
  }
}

// ---------------------------------------------------------------------------
// Pair feature build (bf16 out): feat[p] = [e1+e2, e1*e2, e1, e2]
// ---------------------------------------------------------------------------
__global__ __launch_bounds__(256) void build_feat_kernel(
    const float* __restrict__ x2, const int* __restrict__ i0,
    const int* __restrict__ i1, __bf16* __restrict__ feat) {
  int p = blockIdx.x, t = threadIdx.x;
  int a = i0[p], b = i1[p];
  float e1 = x2[(size_t)a * FD + t];
  float e2 = x2[(size_t)b * FD + t];
  __bf16* f = feat + (size_t)p * (4 * FD);
  f[t]          = (__bf16)(e1 + e2);
  f[FD + t]     = (__bf16)(e1 * e2);
  f[2 * FD + t] = (__bf16)e1;
  f[3 * FD + t] = (__bf16)e2;
}

// ---------------------------------------------------------------------------
// Final matvec: log[p] = RT[p, :512] . dec2_w + dec2_b   (one wave per row)
// ---------------------------------------------------------------------------
__global__ __launch_bounds__(256) void dec2_kernel(
    const __bf16* __restrict__ RT, const float* __restrict__ w,
    const float* __restrict__ b, float* __restrict__ out, int rows) {
  __shared__ float ws[D1D];
  int t = threadIdx.x;
  ws[t] = w[t]; ws[256 + t] = w[256 + t];
  __syncthreads();
  int wave = t >> 5, lane = t & 31;
  int p = blockIdx.x * 8 + wave;
  if (p >= rows) return;
  const __bf16* r = RT + (size_t)p * D1D;
  float s = 0.f;
#pragma unroll
  for (int i = 0; i < 16; ++i) { int c = lane * 16 + i; s += (float)r[c] * ws[c]; }
#pragma unroll
  for (int off = 16; off > 0; off >>= 1) s += __shfl_down(s, off, 32);
  if (lane == 0) out[p] = s + b[0];
}

// ---------------------------------------------------------------------------
// Host launcher
// ---------------------------------------------------------------------------
extern "C" void kernel_launch(void* const* d_in, const int* in_sizes, int n_in,
                              void* d_out, int out_size, void* d_ws, size_t ws_size,
                              hipStream_t stream) {
  (void)in_sizes; (void)n_in; (void)out_size; (void)ws_size;

  // ---- inputs (setup_inputs order) ----
  const float* x_o   = (const float*)d_in[0];
  const float* x_a   = (const float*)d_in[1];
  const int*   ei_o  = (const int*)d_in[2];   // [2, E] : src row 0, dst row 1
  const int*   ei_s  = (const int*)d_in[3];
  const int*   idx   = (const int*)d_in[4];   // [2, P]
  const float* eps_o1 = (const float*)d_in[5];
  const float* w1_o1  = (const float*)d_in[6];
  const float* b1_o1  = (const float*)d_in[7];
  const float* w2_o1  = (const float*)d_in[8];
  const float* b2_o1  = (const float*)d_in[9];
  const float* eps_s1 = (const float*)d_in[10];
  const float* w1_s1  = (const float*)d_in[11];
  const float* b1_s1  = (const float*)d_in[12];
  const float* w2_s1  = (const float*)d_in[13];
  const float* b2_s1  = (const float*)d_in[14];
  const float* eps_o2 = (const float*)d_in[15];
  const float* w1_o2  = (const float*)d_in[16];
  const float* b1_o2  = (const float*)d_in[17];
  const float* w2_o2  = (const float*)d_in[18];
  const float* b2_o2  = (const float*)d_in[19];
  const float* eps_s2 = (const float*)d_in[20];
  const float* w1_s2  = (const float*)d_in[21];
  const float* b1_s2  = (const float*)d_in[22];
  const float* w2_s2  = (const float*)d_in[23];
  const float* b2_s2  = (const float*)d_in[24];
  const float* disc_w = (const float*)d_in[25];
  const float* disc_b = (const float*)d_in[26];
  const float* dec1_w = (const float*)d_in[27]; // [1024, 512]
  const float* dec1_b = (const float*)d_in[28];
  const float* dec2_w = (const float*)d_in[29]; // [512, 1]
  const float* dec2_b = (const float*)d_in[30];

  // ---- output segments (return order: log, ret_os, ret_os_a, x2_os) ----
  float* out_log     = (float*)d_out;            // [P]
  float* out_ret_os  = out_log + PP;             // [N,2]
  float* out_ret_osa = out_ret_os + 2 * NN;      // [N,2]
  float* out_x2      = out_ret_osa + 2 * NN;     // [N,256]

  // ---- workspace layout ----
  char* wsb = (char*)d_ws;
  size_t off = 0;
  auto alloc = [&](size_t bytes) -> void* {
    void* p = wsb + off;
    off += (bytes + 255) & ~(size_t)255;
    return p;
  };
  __bf16* w1o1t = (__bf16*)alloc(256 * 256 * 2);
  __bf16* w2o1t = (__bf16*)alloc(128 * 256 * 2);
  __bf16* w1s1t = (__bf16*)alloc(256 * 256 * 2);
  __bf16* w2s1t = (__bf16*)alloc(128 * 256 * 2);
  __bf16* w1o2t = (__bf16*)alloc(256 * 256 * 2);
  __bf16* w2o2t = (__bf16*)alloc(128 * 256 * 2);
  __bf16* w1s2t = (__bf16*)alloc(256 * 256 * 2);
  __bf16* w2s2t = (__bf16*)alloc(128 * 256 * 2);
  __bf16* dec1t = (__bf16*)alloc((size_t)512 * 1024 * 2);
  float* cs     = (float*)alloc(512 * 4);        // colsum_os | colsum_osa
  float* v_os   = (float*)alloc(256 * 4);
  float* v_osa  = (float*)alloc(256 * 4);
  float*  H    = (float*)alloc((size_t)NN * FD * 4);     // f32: atomic target
  __bf16* T    = (__bf16*)alloc((size_t)NN * FD * 2);    // GIN hidden (bf16)
  __bf16* X1o  = (__bf16*)alloc((size_t)NN * FD * 2);    // layer-1 concat (bf16)
  __bf16* X1a  = (__bf16*)alloc((size_t)NN * FD * 2);
  float*  X2a  = (float*)alloc((size_t)NN * FD * 4);     // f32: readout/disc
  __bf16* FEAT = (__bf16*)alloc((size_t)PP * 4 * FD * 2);
  __bf16* RT   = (__bf16*)alloc((size_t)PP * D1D * 2);

  // ---- weight transpose + bf16 convert ----
  auto xpose = [&](const float* W, __bf16* Wt, int K, int Nout) {
    int n = K * Nout;
    transpose_bf16_kernel<<<(n + 255) / 256, 256, 0, stream>>>(W, Wt, K, Nout);
  };
  xpose(w1_o1, w1o1t, 256, 256); xpose(w2_o1, w2o1t, 256, 128);
  xpose(w1_s1, w1s1t, 256, 256); xpose(w2_s1, w2s1t, 256, 128);
  xpose(w1_o2, w1o2t, 256, 256); xpose(w2_o2, w2o2t, 256, 128);
  xpose(w1_s2, w1s2t, 256, 256); xpose(w2_s2, w2s2t, 256, 128);
  xpose(dec1_w, dec1t, 1024, 512);

  const int NF = NN * FD;
  const int nblkNF = (NF + 255) / 256;

  // ---- layer-1 GIN: f32 node features in, bf16 x1 slab out (outer ReLU) ----
  auto gin1 = [&](const float* X, const int* ei, const float* eps,
                  const __bf16* w1t, const float* b1,
                  const __bf16* w2t, const float* b2, __bf16* Y) {
    scale_copy_kernel<float><<<nblkNF, 256, 0, stream>>>(H, X, eps, NF);
    edge_scatter_kernel<float><<<EE, 256, 0, stream>>>(H, X, ei, ei + EE, EE);
    launch_gemm<true>(H, w1t, b1, T, NN, 256, 256, 256, stream);   // f32 A -> bf16 T
    launch_gemm<true>(T, w2t, b2, Y, NN, 256, 128, 256, stream);   // bf16 A -> bf16 x1
  };

  // ---- layer-2 GIN: bf16 x1 in, f32 x2 slab out (no outer act) ----
  auto gin2 = [&](const __bf16* X, const int* ei, const float* eps,
                  const __bf16* w1t, const float* b1,
                  const __bf16* w2t, const float* b2, float* Y) {
    scale_copy_kernel<__bf16><<<nblkNF, 256, 0, stream>>>(H, X, eps, NF);
    edge_scatter_kernel<__bf16><<<EE, 256, 0, stream>>>(H, X, ei, ei + EE, EE);
    launch_gemm<true>(H, w1t, b1, T, NN, 256, 256, 256, stream);   // f32 A -> bf16 T
    launch_gemm<false>(T, w2t, b2, Y, NN, 256, 128, 256, stream);  // bf16 A -> f32 x2
  };

  auto encode = [&](const float* x, __bf16* x1, float* x2) {
    gin1(x,  ei_o, eps_o1, w1o1t, b1_o1, w2o1t, b2_o1, x1);
    gin1(x,  ei_s, eps_s1, w1s1t, b1_s1, w2s1t, b2_s1, x1 + 128);
    gin2(x1, ei_o, eps_o2, w1o2t, b1_o2, w2o2t, b2_o2, x2);
    gin2(x1, ei_s, eps_s2, w1s2t, b1_s2, w2s2t, b2_s2, x2 + 128);
  };

  encode(x_o, X1o, out_x2);   // x2_os written straight into d_out
  encode(x_a, X1a, X2a);

  // ---- readout + discriminator ----
  zero_kernel<<<2, 256, 0, stream>>>(cs, 512);
  colsum_kernel<<<256, 256, 0, stream>>>(out_x2, cs,       NN);
  colsum_kernel<<<256, 256, 0, stream>>>(X2a,    cs + 256, NN);
  disc_v_kernel<<<1, 256, 0, stream>>>(cs, cs + 256, disc_w, v_os, v_osa,
                                       1.0f / (float)NN);
  disc_scores_kernel<<<(NN + 7) / 8, 256, 0, stream>>>(
      out_x2, X2a, v_os, v_osa, disc_b, out_ret_os, out_ret_osa, NN);

  // ---- link decoder ----
  build_feat_kernel<<<PP, 256, 0, stream>>>(out_x2, idx, idx + PP, FEAT);
  launch_gemm<true>(FEAT, dec1t, dec1_b, RT, PP, 1024, 512, 512, stream); // bf16 A
  dec2_kernel<<<(PP + 7) / 8, 256, 0, stream>>>(RT, dec2_w, dec2_b, out_log, PP);
}
